// MultiheadEnergyAttention_32255204393271
// MI455X (gfx1250) — matrix operand order repaired and checked by reading, not compiled, not run on hardware
//
#include <hip/hip_runtime.h>

// ---------------------------------------------------------------------------
// MultiheadEnergyAttention for MI455X (gfx1250, wave32, WMMA bf16).
//   B=2, N=2048, D=1024, H=16, Z=64.
// Pipeline:
//   1) cvt_bf16:   x, wk, wq  f32 -> bf16 in workspace
//   2) proj_gemm:  K = Xb @ Wkb^T, Q = Xb @ Wqb^T   (2-step lookahead software
//                  pipeline, fully unrolled, v_wmma_f32_16x16x32_bf16)
//   3) attn_lse:   8 waves/block share one (b,h); key stream staged through a
//                  double-buffered LDS chunk (128 rows x 64 z, bf16); online
//                  logsumexp in registers; scores never touch memory
//   4) reduce:     fixed-order tree sum of 4096 partials -> scalar energy
// ---------------------------------------------------------------------------

typedef __attribute__((ext_vector_type(16))) __bf16 v16bf;
typedef __attribute__((ext_vector_type(8)))  __bf16 v8bf;
typedef __attribute__((ext_vector_type(8)))  float  v8f;

#define BATCH 2
#define SEQ   2048
#define EMBED 1024
#define HEADS 16
#define HDIM  64

#define KSTEPS      (EMBED / 32)            // 32 K-steps in the projections
#define CHUNK_TILES 8                       // key tiles per LDS chunk
#define CHUNK_ROWS  (CHUNK_TILES * 16)      // 128 key rows
#define NCHUNKS     (SEQ / CHUNK_ROWS)      // 16

// A-fragment (16x32 bf16, M x K). ISA 7.12.2: lane L holds row M=L%16;
// VGPR0-3 -> K = hi*8 + 0..7, VGPR4-7 -> K = 16 + hi*8 + 0..7 (hi = L/16).
__device__ __forceinline__ v16bf load_afrag(const __bf16* p, int hi) {
  v8bf lo = *(const v8bf*)(p + hi * 8);
  v8bf up = *(const v8bf*)(p + 16 + hi * 8);
  return __builtin_shufflevector(lo, up, 0,1,2,3,4,5,6,7,8,9,10,11,12,13,14,15);
}

// B-fragment (32x16 bf16, K x N). Lane L holds col N=L%16;
// VGPR0-3 -> K = hi*8 + 0..7, VGPR4-7 -> K = 16 + hi*8 + 0..7 (hi = L/16).
__device__ __forceinline__ v16bf load_bfrag(const __bf16* p, int hi) {
  v8bf lo = *(const v8bf*)(p + hi * 8);
  v8bf up = *(const v8bf*)(p + 16 + hi * 8);
  return __builtin_shufflevector(lo, up, 0,1,2,3,4,5,6,7,8,9,10,11,12,13,14,15);
}

// ---------------------------------------------------------------- 1) convert
__global__ void __launch_bounds__(256)
cvt_bf16(const float* __restrict__ x, const float* __restrict__ wk,
         const float* __restrict__ wq, __bf16* __restrict__ xb,
         __bf16* __restrict__ wkb, __bf16* __restrict__ wqb) {
  const int NX = BATCH * SEQ * EMBED;
  const int NW = HEADS * HDIM * EMBED;
  const int total = NX + 2 * NW;
  for (int i = blockIdx.x * blockDim.x + threadIdx.x; i < total;
       i += gridDim.x * blockDim.x) {
    if (i < NX)            xb[i]            = (__bf16)x[i];
    else if (i < NX + NW)  wkb[i - NX]      = (__bf16)wk[i - NX];
    else                   wqb[i - NX - NW] = (__bf16)wq[i - NX - NW];
  }
}

// ------------------------------------------------------------ 2) projections
// C[4096 x 1024] = Xb[4096 x 1024] * Wb[1024 x 1024]^T.
// One wave -> 16(M) x 64(N) strip. 2-step lookahead: the loads for K-step s+2
// are issued right after the WMMAs of step s, so 4 WMMAs separate each load
// from its consumer in program order. Fully unrolled -> slot rotation is pure
// SSA renaming (no v_mov copies).
__global__ void __launch_bounds__(128)
proj_gemm(const __bf16* __restrict__ xb, const __bf16* __restrict__ wb,
          __bf16* __restrict__ outb) {
  const int lane = threadIdx.x & 31;
  const int wave = threadIdx.x >> 5;
  const int tile = blockIdx.x * 4 + wave;      // 4096 tiles total
  const int m0 = (tile & 255) * 16;            // 256 M-tiles
  const int n0 = (tile >> 8) * 64;             // 16 N-strips of 64
  const int ln = lane & 15, hi = lane >> 4;

  const __bf16* arow = xb + (size_t)(m0 + ln) * EMBED;
  const __bf16* brow = wb + (size_t)(n0 + ln) * EMBED;

  v8f acc[4] = {};
  v16bf aS[2];
  v16bf bS[2][4];
#pragma unroll
  for (int s = 0; s < 2; ++s) {                // prologue: steps 0 and 1
    aS[s] = load_afrag(arow + s * 32, hi);
#pragma unroll
    for (int j = 0; j < 4; ++j)
      bS[s][j] = load_bfrag(brow + (size_t)j * 16 * EMBED + s * 32, hi);
  }

#pragma unroll
  for (int s = 0; s < KSTEPS; ++s) {
    const int slot = s & 1;
#pragma unroll
    for (int j = 0; j < 4; ++j)                // consume step s
      acc[j] = __builtin_amdgcn_wmma_f32_16x16x32_bf16(
          false, aS[slot], false, bS[slot][j], (short)0, acc[j], false, false);
    if (s + 2 < KSTEPS) {                      // refill slot with step s+2
      const int kk = (s + 2) * 32;
      aS[slot] = load_afrag(arow + kk, hi);
#pragma unroll
      for (int j = 0; j < 4; ++j)
        bS[slot][j] = load_bfrag(brow + (size_t)j * 16 * EMBED + kk, hi);
    }
  }

  // C layout: VGPR r, lane L -> row (r + 8*hi), col (L%16)
#pragma unroll
  for (int j = 0; j < 4; ++j)
#pragma unroll
    for (int r = 0; r < 8; ++r)
      outb[(size_t)(m0 + r + 8 * hi) * EMBED + n0 + j * 16 + ln] =
          (__bf16)acc[j][r];
}

// ------------------------------------------------- 3) scores + online LSE
// Block = 256 threads = 8 waves, all on one (b,h); wave w owns query rows
// [qblk*128 + w*16, +16). Key stream staged through double-buffered LDS:
// chunk c+1 is prefetched to registers while chunk c is consumed from LDS.
__global__ void __launch_bounds__(256)
attn_lse(const __bf16* __restrict__ qb, const __bf16* __restrict__ kb,
         const float* __restrict__ betas, float* __restrict__ partial) {
  __shared__ __align__(16) __bf16 kshare[2][CHUNK_ROWS * HDIM];  // 2 x 16 KB

  const int tid  = threadIdx.x;
  const int lane = tid & 31;
  const int wave = tid >> 5;
  const int qblk = blockIdx.x & 15;            // 16 query blocks per (b,h)
  const int bh   = blockIdx.x >> 4;            // 0..31
  const int h    = bh & 15;
  const int bidx = bh >> 4;
  const int qtile = qblk * 8 + wave;           // 0..127
  const int n0 = qtile * 16;
  const int ln = lane & 15, hi = lane >> 4;
  const float beta = betas[h];

  const __bf16* qrow = qb + (size_t)(bidx * SEQ + n0 + ln) * EMBED + h * HDIM;
  const v16bf q0 = load_afrag(qrow, hi);       // z = 0..31
  const v16bf q1 = load_afrag(qrow + 32, hi);  // z = 32..63

  const __bf16* kbase = kb + (size_t)bidx * SEQ * EMBED + h * HDIM;

  float mx[8], sm[8];
#pragma unroll
  for (int r = 0; r < 8; ++r) { mx[r] = -3.0e38f; sm[r] = 0.0f; }

  // Each thread stages 4 x b128 units per chunk: key row = tid>>3 + 32*i,
  // z offset = (tid&7)*8.
  const int srow = tid >> 3;
  const int szo  = (tid & 7) * 8;
  v8bf stage[4];
#pragma unroll
  for (int i = 0; i < 4; ++i)                  // prologue: chunk 0 -> regs
    stage[i] = *(const v8bf*)(kbase + (size_t)(srow + 32 * i) * EMBED + szo);
#pragma unroll
  for (int i = 0; i < 4; ++i)                  // regs -> LDS buffer 0
    *(v8bf*)&kshare[0][(srow + 32 * i) * HDIM + szo] = stage[i];
  __syncthreads();

#pragma unroll 2
  for (int c = 0; c < NCHUNKS; ++c) {
    if (c + 1 < NCHUNKS) {                     // prefetch chunk c+1 -> regs
      const __bf16* src = kbase + (size_t)(c + 1) * CHUNK_ROWS * EMBED;
#pragma unroll
      for (int i = 0; i < 4; ++i)
        stage[i] = *(const v8bf*)(src + (size_t)(srow + 32 * i) * EMBED + szo);
    }
    const __bf16* ks = &kshare[c & 1][0];
#pragma unroll
    for (int kt = 0; kt < CHUNK_TILES; ++kt) { // consume chunk c from LDS
      const __bf16* krow = ks + (kt * 16 + ln) * HDIM;
      v16bf k0 = load_bfrag(krow, hi);         // ds_load_b128 x2
      v16bf k1 = load_bfrag(krow + 32, hi);
      v8f acc = {};
      acc = __builtin_amdgcn_wmma_f32_16x16x32_bf16(
          false, q0, false, k0, (short)0, acc, false, false);
      acc = __builtin_amdgcn_wmma_f32_16x16x32_bf16(
          false, q1, false, k1, (short)0, acc, false, false);
      const int mcol = c * CHUNK_ROWS + kt * 16 + ln;
#pragma unroll
      for (int r = 0; r < 8; ++r) {
        float s = beta * acc[r];
        if (n0 + r + 8 * hi == mcol) s = -1.0e9f;   // mask self-pair
        float nm = fmaxf(mx[r], s);
        sm[r] = sm[r] * __expf(mx[r] - nm) + __expf(s - nm);
        mx[r] = nm;
      }
    }
    __syncthreads();                           // all waves done with chunk c
    if (c + 1 < NCHUNKS) {
#pragma unroll
      for (int i = 0; i < 4; ++i)              // regs -> other LDS buffer
        *(v8bf*)&kshare[(c + 1) & 1][(srow + 32 * i) * HDIM + szo] = stage[i];
      __syncthreads();
    }
  }

  // Merge 16 lanes of each column group (lane L holds cols m%16==L%16),
  // then sum lse over this wave's 16 rows.
  float energy = 0.0f;
#pragma unroll
  for (int r = 0; r < 8; ++r) {
    float m = mx[r], s = sm[r];
#pragma unroll
    for (int mask = 1; mask < 16; mask <<= 1) {
      float mo = __shfl_xor(m, mask, 32);
      float so = __shfl_xor(s, mask, 32);
      float nm = fmaxf(m, mo);
      s = s * __expf(m - nm) + so * __expf(mo - nm);
      m = nm;
    }
    energy += m + __logf(s);                   // lse for row r + 8*hi
  }
  energy += __shfl_xor(energy, 16, 32);        // rows 0..7 + rows 8..15
  if (lane == 0) partial[bh * 128 + qtile] = -energy / beta;
}

// ----------------------------------------------------------- 4) reduction
__global__ void __launch_bounds__(256)
reduce_energy(const float* __restrict__ partial, float* __restrict__ out) {
  __shared__ float buf[256];
  float s = 0.0f;
  for (int i = threadIdx.x; i < 4096; i += 256) s += partial[i];
  buf[threadIdx.x] = s;
  __syncthreads();
  for (int step = 128; step > 0; step >>= 1) {
    if ((int)threadIdx.x < step) buf[threadIdx.x] += buf[threadIdx.x + step];
    __syncthreads();
  }
  if (threadIdx.x == 0) out[0] = buf[0] / (float)(BATCH * SEQ);
}

// ---------------------------------------------------------------------------
extern "C" void kernel_launch(void* const* d_in, const int* in_sizes, int n_in,
                              void* d_out, int out_size, void* d_ws,
                              size_t ws_size, hipStream_t stream) {
  const float* x     = (const float*)d_in[0];
  const float* wk    = (const float*)d_in[1];
  const float* wq    = (const float*)d_in[2];
  const float* betas = (const float*)d_in[3];
  float* out = (float*)d_out;

  // Workspace layout (bytes); total ~28.0 MB.
  char* ws = (char*)d_ws;
  __bf16* xb      = (__bf16*)(ws);                 //  8,388,608
  __bf16* wkb     = (__bf16*)(ws + 8388608);       //  2,097,152
  __bf16* wqb     = (__bf16*)(ws + 10485760);      //  2,097,152
  __bf16* kbuf    = (__bf16*)(ws + 12582912);      //  8,388,608
  __bf16* qbuf    = (__bf16*)(ws + 20971520);      //  8,388,608
  float*  partial = (float*)(ws + 29360128);       //     16,384

  cvt_bf16<<<2048, 256, 0, stream>>>(x, wk, wq, xb, wkb, wqb);
  proj_gemm<<<1024, 128, 0, stream>>>(xb, wkb, kbuf);   // K projection
  proj_gemm<<<1024, 128, 0, stream>>>(xb, wqb, qbuf);   // Q projection
  attn_lse<<<512, 256, 0, stream>>>(qbuf, kbuf, betas, partial);
  reduce_energy<<<1, 256, 0, stream>>>(partial, out);
}
